// DualGraphSHM_26645977104430
// MI455X (gfx1250) — compile-verified, matrix operand order checked by reading, not run on hardware
//
#include <hip/hip_runtime.h>
#include <hip/hip_bf16.h>
#include <math.h>

typedef __attribute__((ext_vector_type(16))) _Float16 v16h;
typedef __attribute__((ext_vector_type(8)))  float    v8f;

#define NTHREADS 256
#define NWAVES   8

// ---------------------------------------------------------------------------
// WMMA helpers (v_wmma_f32_16x16x32_f16, wave32)
// A 16x32 f16: lane holds row M=lane&15; K elements per lane-half:
//   lanes 0-15 -> K = {0..7, 16..23}, lanes 16-31 -> K = {8..15, 24..31}
// C/D 16x16 f32: VGPR v, lane l -> M = v + 8*(l>>4), N = l&15
// Fragment elements are two contiguous 8-element runs in K, so any source
// that is contiguous along K vectorizes into b128 loads.
// ---------------------------------------------------------------------------
__device__ __forceinline__ int frag_koff(int e, int lane) {
  int khalf = ((lane >> 4) << 3);
  return (e < 8) ? (khalf + e) : (16 + khalf + (e - 8));
}

// Generic M=32 (2 tiles) x Ntiles*16 GEMM, K = Kp (multiple of 32).
// ag(m,k) -> _Float16 A element; bg(k,n) -> _Float16 B element;
// ep(M,N,val) epilogue (guards/bias/activation/store).
template <class AG, class BG, class EP>
__device__ __forceinline__ void wmma_gemm(AG ag, BG bg, int Kp, int Ntiles, EP ep) {
  const int wave = threadIdx.x >> 5;
  const int lane = threadIdx.x & 31;
  const int row  = lane & 15;
  const int mhi  = (lane >> 4) << 3;
  const int ntile = 2 * Ntiles;
  for (int t = wave; t < ntile; t += NWAVES) {
    const int m0 = (t & 1) << 4;
    const int n0 = (t >> 1) << 4;
    v8f acc = {};
    for (int k0 = 0; k0 < Kp; k0 += 32) {
      v16h a, bf;
#pragma unroll
      for (int e = 0; e < 16; ++e) {
        int kk = k0 + frag_koff(e, lane);
        a[e]  = ag(m0 + row, kk);
        bf[e] = bg(kk, n0 + row);
      }
      acc = __builtin_amdgcn_wmma_f32_16x16x32_f16(
          /*neg_a=*/false, a, /*neg_b=*/false, bf,
          /*c_mod=*/(short)0, acc, /*reuse_a=*/false, /*reuse_b=*/false);
    }
#pragma unroll
    for (int v = 0; v < 8; ++v) ep(m0 + v + mhi, n0 + row, acc[v]);
  }
}

__device__ __forceinline__ float sigmoidf_(float x) { return 1.0f / (1.0f + expf(-x)); }

// ---------------------------------------------------------------------------
// Kernel 0: pad+cast fp32 weight [K,N] -> TRANSPOSED f16 [Np][Kp] (zero pad).
// Transposed so WMMA B-fragments (fixed n, contiguous k runs) vectorize into
// global_load_b128 instead of 16 scalar d16 loads + v_mov_b16 packing.
// ---------------------------------------------------------------------------
__global__ void pad_cast_f16T(const float* __restrict__ src, _Float16* __restrict__ dst,
                              int K, int N, int Kp, int Np) {
  int idx = blockIdx.x * blockDim.x + threadIdx.x;
  int total = Kp * Np;
  if (idx >= total) return;
  int n = idx / Kp, k = idx % Kp;  // dst[n][k]
  dst[idx] = (k < K && n < N) ? (_Float16)src[k * N + n] : (_Float16)0.0f;
}

// ---------------------------------------------------------------------------
// Kernel 1: per-batch fused conv1+conv2 (VALU, LDS-chunked) then WMMA GEMM
//   F2[b] = relu(h2 @ Wt + bt)   [30,1024] x [1024,200(pad208)]
// ---------------------------------------------------------------------------
__global__ __launch_bounds__(NTHREADS) void conv_proj_kernel(
    const float* __restrict__ x, const float* __restrict__ Wc1,
    const float* __restrict__ bc1, const float* __restrict__ Wc2,
    const float* __restrict__ bc2, const _Float16* __restrict__ WtT,
    const float* __restrict__ bt, float* __restrict__ F2) {
  extern __shared__ char smem[];
  float* cw1 = (float*)smem;            // 2700
  float* cw2 = cw1 + 2700;              // 2700
  float* b1s = cw2 + 2700;              // 32
  float* b2s = b1s + 32;                // 32
  float* xch = b2s + 32;                // 30*260
  float* h1c = xch + 30 * 260;          // 30*258
  _Float16* h2 = (_Float16*)(h1c + 30 * 258);  // 32*1024

  const int tid = threadIdx.x;
  const int b = blockIdx.x;
  const float* xb = x + (size_t)b * 30 * 1024;

  for (int i = tid; i < 2700; i += NTHREADS) { cw1[i] = Wc1[i]; cw2[i] = Wc2[i]; }
  for (int i = tid; i < 32; i += NTHREADS) {
    b1s[i] = (i < 30) ? bc1[i] : 0.0f;
    b2s[i] = (i < 30) ? bc2[i] : 0.0f;
  }
  for (int i = tid; i < 32 * 1024; i += NTHREADS) h2[i] = (_Float16)0.0f;
  __syncthreads();

  for (int tc = 0; tc < 1024; tc += 256) {
    // stage x with halo: column j -> global t = tc + j - 2
    for (int i = tid; i < 30 * 260; i += NTHREADS) {
      int s = i / 260, j = i % 260;
      int tg = tc + j - 2;
      xch[i] = (tg >= 0 && tg < 1024) ? xb[s * 1024 + tg] : 0.0f;
    }
    __syncthreads();
    // conv1 (SAME, k=3): h1 column j -> global t = tc + j - 1
    for (int i = tid; i < 30 * 258; i += NTHREADS) {
      int s = i / 258, j = i % 258;
      float acc = b1s[s];
      for (int s2 = 0; s2 < 30; ++s2) {
        const float* w = cw1 + (s * 30 + s2) * 3;
        const float* xp = xch + s2 * 260 + j;
        acc += xp[0] * w[0] + xp[1] * w[1] + xp[2] * w[2];
      }
      h1c[i] = fmaxf(acc, 0.0f);
    }
    __syncthreads();
    // conv2: h2[s][tc+u]
    for (int i = tid; i < 30 * 256; i += NTHREADS) {
      int s = i / 256, u = i % 256;
      float acc = b2s[s];
      for (int s2 = 0; s2 < 30; ++s2) {
        const float* w = cw2 + (s * 30 + s2) * 3;
        const float* hp = h1c + s2 * 258 + u;
        acc += hp[0] * w[0] + hp[1] * w[1] + hp[2] * w[2];
      }
      h2[s * 1024 + tc + u] = (_Float16)fmaxf(acc, 0.0f);
    }
    __syncthreads();
  }

  // F2[b] = relu(h2 @ Wt + bt): M=32(valid 30), K=1024, N=208(valid 200)
  float* F2b = F2 + (size_t)b * 30 * 200;
  wmma_gemm(
      [&](int m, int k) -> _Float16 { return h2[m * 1024 + k]; },
      [&](int k, int n) -> _Float16 { return WtT[n * 1024 + k]; },
      1024, 13,
      [&](int M, int N, float v) {
        if (M < 30 && N < 200) F2b[M * 200 + N] = fmaxf(v + bt[N], 0.0f);
      });
}

// ---------------------------------------------------------------------------
// Kernel 2: per-batch graph pipeline, all intermediates in LDS (zero-padded
// [32][224] fp32 buffers), every contraction through wmma_gemm.
// All f16 weights are transposed [n][224] so B-fragments vectorize.
// ---------------------------------------------------------------------------
struct GraphArgs {
  const float* F2g;
  const float* adj;
  const _Float16 *Wa, *Wm1, *Wm2, *Wm3, *Wg1, *Wg2, *Wl, *Wgl, *Wp0, *Wp1, *Wp2;
  const float *wg, *Ws1, *Ws2, *Wf1, *Wf2, *Wcls, *bcls;
  float* out;
};

#define LD2 224
#define LDG 304
#define POOLF 65416  // floats

__global__ __launch_bounds__(NTHREADS) void graph_chain_kernel(GraphArgs A) {
  extern __shared__ char smem[];
  float* pool  = (float*)smem;
  float* adjf  = pool;               // 32*32
  float* AF    = adjf + 1024;        // 32*32
  float* F2    = AF + 1024;          // 32*224
  float* S0    = F2 + 32 * LD2;      // scratch 32*224
  float* X1    = S0 + 32 * LD2;
  float* X2    = X1 + 32 * LD2;
  float* X3    = X2 + 32 * LD2;
  float* XS    = X3 + 32 * LD2;
  float* GH    = XS + 32 * LD2;      // 32*304
  float* GV    = GH + 32 * LDG;      // 32*304
  float* gvals = GV + 32 * LDG;      // 32*3
  float* wch   = gvals + 96;         // 32
  float* mbuf  = wch + 32;           // 304
  float* u2    = mbuf + 304;         // 160
  float* wft   = u2 + 160;           // 304
  float* lbuf  = wft + 304;          // 8

  const int tid = threadIdx.x;
  const int b = blockIdx.x;

  for (int i = tid; i < POOLF; i += NTHREADS) pool[i] = 0.0f;
  __syncthreads();
  for (int i = tid; i < 30 * 30; i += NTHREADS) adjf[(i / 30) * 32 + (i % 30)] = A.adj[i];
  const float* F2src = A.F2g + (size_t)b * 6000;
  for (int i = tid; i < 6000; i += NTHREADS) F2[(i / 200) * LD2 + (i % 200)] = F2src[i];
  __syncthreads();

  // ---- A_F = softmax(relu(F2 @ Wa @ F2^T)) ----
  wmma_gemm([&](int m, int k) -> _Float16 { return (_Float16)F2[m * LD2 + k]; },
            [&](int k, int n) -> _Float16 { return A.Wa[n * LD2 + k]; }, 224, 13,
            [&](int M, int N, float v) { if (M < 30 && N < 200) S0[M * LD2 + N] = v; });
  __syncthreads();
  wmma_gemm([&](int m, int k) -> _Float16 { return (_Float16)S0[m * LD2 + k]; },
            [&](int k, int n) -> _Float16 { return (_Float16)F2[n * LD2 + k]; }, 224, 2,
            [&](int M, int N, float v) { if (M < 30 && N < 30) AF[M * 32 + N] = fmaxf(v, 0.0f); });
  __syncthreads();
  if (tid < 30) {
    float m = -1e30f;
    for (int j = 0; j < 30; ++j) m = fmaxf(m, AF[tid * 32 + j]);
    float s = 0.0f;
    for (int j = 0; j < 30; ++j) { float e = expf(AF[tid * 32 + j] - m); AF[tid * 32 + j] = e; s += e; }
    float inv = 1.0f / s;
    for (int j = 0; j < 30; ++j) AF[tid * 32 + j] *= inv;
  }
  __syncthreads();

  // ---- signal GCN: h1 = relu(adj@F2@Wg1); xs = adj@h1@Wg2 ----
  wmma_gemm([&](int m, int k) -> _Float16 { return (_Float16)adjf[m * 32 + k]; },
            [&](int k, int n) -> _Float16 { return (_Float16)F2[k * LD2 + n]; }, 32, 13,
            [&](int M, int N, float v) { if (M < 30 && N < 200) S0[M * LD2 + N] = v; });
  __syncthreads();
  wmma_gemm([&](int m, int k) -> _Float16 { return (_Float16)S0[m * LD2 + k]; },
            [&](int k, int n) -> _Float16 { return A.Wg1[n * LD2 + k]; }, 224, 13,
            [&](int M, int N, float v) { if (M < 30 && N < 200) XS[M * LD2 + N] = fmaxf(v, 0.0f); });
  __syncthreads();
  wmma_gemm([&](int m, int k) -> _Float16 { return (_Float16)adjf[m * 32 + k]; },
            [&](int k, int n) -> _Float16 { return (_Float16)XS[k * LD2 + n]; }, 32, 13,
            [&](int M, int N, float v) { if (M < 30 && N < 200) S0[M * LD2 + N] = v; });
  __syncthreads();
  wmma_gemm([&](int m, int k) -> _Float16 { return (_Float16)S0[m * LD2 + k]; },
            [&](int k, int n) -> _Float16 { return A.Wg2[n * LD2 + k]; }, 224, 13,
            [&](int M, int N, float v) { if (M < 30 && N < 200) XS[M * LD2 + N] = v; });
  __syncthreads();

  // ---- MGCN: x1,x2,x3 ----
  const _Float16* Wm[3] = {A.Wm1, A.Wm2, A.Wm3};
  float* Xo[3] = {X1, X2, X3};
  const float* Xin = F2;
  for (int L = 0; L < 3; ++L) {
    wmma_gemm([&](int m, int k) -> _Float16 { return (_Float16)AF[m * 32 + k]; },
              [&](int k, int n) -> _Float16 { return (_Float16)Xin[k * LD2 + n]; }, 32, 13,
              [&](int M, int N, float v) { if (M < 30 && N < 200) S0[M * LD2 + N] = v; });
    __syncthreads();
    const _Float16* W = Wm[L];
    float* Xd = Xo[L];
    wmma_gemm([&](int m, int k) -> _Float16 { return (_Float16)S0[m * LD2 + k]; },
              [&](int k, int n) -> _Float16 { return W[n * LD2 + k]; }, 224, 13,
              [&](int M, int N, float v) { if (M < 30 && N < 200) Xd[M * LD2 + N] = fmaxf(v, 0.0f); });
    __syncthreads();
    Xin = Xo[L];
  }

  // ---- Hk = (xk + xs) * 0.5 (in place) ----
  for (int i = tid; i < 6000; i += NTHREADS) {
    int idx = (i / 200) * LD2 + (i % 200);
    float xs = XS[idx];
    X1[idx] = (X1[idx] + xs) * 0.5f;
    X2[idx] = (X2[idx] + xs) * 0.5f;
    X3[idx] = (X3[idx] + xs) * 0.5f;
  }
  __syncthreads();

  // ---- gating over 3 branches ----
  if (tid < 30) {
    float sc[3];
    const float* Hs[3] = {X1 + tid * LD2, X2 + tid * LD2, X3 + tid * LD2};
    for (int k = 0; k < 3; ++k) {
      float a = 0.0f;
      for (int f = 0; f < 200; ++f) a += Hs[k][f] * A.wg[f * 3 + k];
      sc[k] = a;
    }
    float m = fmaxf(sc[0], fmaxf(sc[1], sc[2]));
    float e0 = expf(sc[0] - m), e1 = expf(sc[1] - m), e2 = expf(sc[2] - m);
    float inv = 1.0f / (e0 + e1 + e2);
    gvals[tid * 3 + 0] = e0 * inv;
    gvals[tid * 3 + 1] = e1 * inv;
    gvals[tid * 3 + 2] = e2 * inv;
  }
  __syncthreads();

  // ---- G_h = concat_k( adj @ (g_k * Hk) @ Wpk ) -> [30,300] ----
  const _Float16* Wpk[3] = {A.Wp0, A.Wp1, A.Wp2};
  for (int k = 0; k < 3; ++k) {
    const float* H = Xo[k];
    wmma_gemm([&](int m, int kk) -> _Float16 { return (_Float16)adjf[m * 32 + kk]; },
              [&](int kk, int n) -> _Float16 { return (_Float16)(gvals[kk * 3 + k] * H[kk * LD2 + n]); },
              32, 13,
              [&](int M, int N, float v) { if (M < 30 && N < 200) S0[M * LD2 + N] = v; });
    __syncthreads();
    const _Float16* Wp = Wpk[k];
    wmma_gemm([&](int m, int kk) -> _Float16 { return (_Float16)S0[m * LD2 + kk]; },
              [&](int kk, int n) -> _Float16 { return Wp[n * LD2 + kk]; }, 224, 7,
              [&](int M, int N, float v) { if (M < 30 && N < 100) GH[M * LDG + k * 100 + N] = v; });
    __syncthreads();
  }

  // ---- G_v = concat(loc, glb) -> [30,300] ----
  wmma_gemm([&](int m, int k) -> _Float16 { return (_Float16)adjf[m * 32 + k]; },
            [&](int k, int n) -> _Float16 { return (_Float16)F2[k * LD2 + n]; }, 32, 13,
            [&](int M, int N, float v) { if (M < 30 && N < 200) S0[M * LD2 + N] = v; });
  __syncthreads();
  wmma_gemm([&](int m, int k) -> _Float16 { return (_Float16)S0[m * LD2 + k]; },
            [&](int k, int n) -> _Float16 { return A.Wl[n * LD2 + k]; }, 224, 10,
            [&](int M, int N, float v) { if (M < 30 && N < 150) GV[M * LDG + N] = fmaxf(v, 0.0f); });
  __syncthreads();
  wmma_gemm([&](int m, int k) -> _Float16 { return (_Float16)AF[m * 32 + k]; },
            [&](int k, int n) -> _Float16 { return (_Float16)F2[k * LD2 + n]; }, 32, 13,
            [&](int M, int N, float v) { if (M < 30 && N < 200) S0[M * LD2 + N] = v; });
  __syncthreads();
  wmma_gemm([&](int m, int k) -> _Float16 { return (_Float16)S0[m * LD2 + k]; },
            [&](int k, int n) -> _Float16 { return A.Wgl[n * LD2 + k]; }, 224, 10,
            [&](int M, int N, float v) { if (M < 30 && N < 150) GV[M * LDG + 150 + N] = fmaxf(v, 0.0f); });
  __syncthreads();

  // ---- channel SE: wch = sigmoid(relu(mean_f(Gv) @ Ws1) @ Ws2) ----
  if (tid < 30) {
    float a = 0.0f;
    for (int f = 0; f < 300; ++f) a += GV[tid * LDG + f];
    wch[tid] = a * (1.0f / 300.0f);
  }
  __syncthreads();
  if (tid < 15) {
    float a = 0.0f;
    for (int s = 0; s < 30; ++s) a += wch[s] * A.Ws1[s * 15 + tid];
    mbuf[tid] = fmaxf(a, 0.0f);
  }
  __syncthreads();
  if (tid < 30) {
    float a = 0.0f;
    for (int j = 0; j < 15; ++j) a += mbuf[j] * A.Ws2[j * 30 + tid];
    wch[tid] = sigmoidf_(a);
  }
  __syncthreads();

  // ---- feature SE: wft = sigmoid(relu(mean_s(Gh) @ Wf1) @ Wf2) ----
  for (int f = tid; f < 300; f += NTHREADS) {
    float a = 0.0f;
    for (int s = 0; s < 30; ++s) a += GH[s * LDG + f];
    mbuf[f] = a * (1.0f / 30.0f);
  }
  __syncthreads();
  for (int j = tid; j < 150; j += NTHREADS) {
    float a = 0.0f;
    for (int f = 0; f < 300; ++f) a += mbuf[f] * A.Wf1[f * 150 + j];
    u2[j] = fmaxf(a, 0.0f);
  }
  __syncthreads();
  for (int f = tid; f < 300; f += NTHREADS) {
    float a = 0.0f;
    for (int j = 0; j < 150; ++j) a += u2[j] * A.Wf2[j * 300 + f];
    wft[f] = sigmoidf_(a);
  }
  __syncthreads();

  // ---- classifier: logits = G @ Wcls + bcls; log_softmax ----
  float lacc[7] = {0, 0, 0, 0, 0, 0, 0};
  for (int idx = tid; idx < 18000; idx += NTHREADS) {
    int s = idx / 600, j = idx % 600;
    float gval = (j < 300) ? GH[s * LDG + j] * wch[s]
                           : GV[s * LDG + (j - 300)] * wft[j - 300];
    const float* wc = A.Wcls + (size_t)idx * 7;
    __builtin_prefetch(A.Wcls + (size_t)(idx + NTHREADS) * 7, 0, 0);
#pragma unroll
    for (int c = 0; c < 7; ++c) lacc[c] += gval * wc[c];
  }
  float* red = S0;  // free now; 256*8 floats
  for (int c = 0; c < 7; ++c) red[tid * 8 + c] = lacc[c];
  __syncthreads();
  if (tid < 7) {
    float s = A.bcls[tid];
    for (int t2 = 0; t2 < NTHREADS; ++t2) s += red[t2 * 8 + tid];
    lbuf[tid] = s;
  }
  __syncthreads();
  if (tid == 0) {
    float m = lbuf[0];
    for (int c = 1; c < 7; ++c) m = fmaxf(m, lbuf[c]);
    float s = 0.0f;
    for (int c = 0; c < 7; ++c) s += expf(lbuf[c] - m);
    float ls = logf(s);
    float* o = A.out + (size_t)b * 7;
    for (int c = 0; c < 7; ++c) o[c] = lbuf[c] - m - ls;
  }
}

// ---------------------------------------------------------------------------
extern "C" void kernel_launch(void* const* d_in, const int* in_sizes, int n_in,
                              void* d_out, int out_size, void* d_ws, size_t ws_size,
                              hipStream_t stream) {
  const float* x    = (const float*)d_in[0];
  const float* adj  = (const float*)d_in[1];
  const float* Wc1  = (const float*)d_in[2];
  const float* bc1  = (const float*)d_in[3];
  const float* Wc2  = (const float*)d_in[4];
  const float* bc2  = (const float*)d_in[5];
  const float* Wt   = (const float*)d_in[6];
  const float* bt   = (const float*)d_in[7];
  const float* Wa   = (const float*)d_in[8];
  const float* Wm1  = (const float*)d_in[9];
  const float* Wm2  = (const float*)d_in[10];
  const float* Wm3  = (const float*)d_in[11];
  const float* Wg1  = (const float*)d_in[12];
  const float* Wg2  = (const float*)d_in[13];
  const float* wg   = (const float*)d_in[14];
  const float* Wp1  = (const float*)d_in[15];
  const float* Wp2  = (const float*)d_in[16];
  const float* Wp3  = (const float*)d_in[17];
  const float* Wl   = (const float*)d_in[18];
  const float* Wgl  = (const float*)d_in[19];
  const float* Ws1  = (const float*)d_in[20];
  const float* Ws2  = (const float*)d_in[21];
  const float* Wf1  = (const float*)d_in[22];
  const float* Wf2  = (const float*)d_in[23];
  const float* Wcls = (const float*)d_in[24];
  const float* bcls = (const float*)d_in[25];
  float* out = (float*)d_out;

  const int B = in_sizes[0] / (30 * 1024);

  // ---- workspace carve (aligned 256B) ----
  char* ws = (char*)d_ws;
  size_t off = 0;
  auto carve = [&](size_t bytes) -> char* {
    char* p = ws + off;
    off = (off + bytes + 255) & ~(size_t)255;
    return p;
  };
  // transposed f16 weights: [Np][Kp]
  _Float16* WtT  = (_Float16*)carve((size_t)208 * 1024 * 2);
  _Float16* WaT  = (_Float16*)carve((size_t)208 * 224 * 2);
  _Float16* Wm1T = (_Float16*)carve((size_t)208 * 224 * 2);
  _Float16* Wm2T = (_Float16*)carve((size_t)208 * 224 * 2);
  _Float16* Wm3T = (_Float16*)carve((size_t)208 * 224 * 2);
  _Float16* Wg1T = (_Float16*)carve((size_t)208 * 224 * 2);
  _Float16* Wg2T = (_Float16*)carve((size_t)208 * 224 * 2);
  _Float16* WlT  = (_Float16*)carve((size_t)160 * 224 * 2);
  _Float16* WglT = (_Float16*)carve((size_t)160 * 224 * 2);
  _Float16* Wp1T = (_Float16*)carve((size_t)112 * 224 * 2);
  _Float16* Wp2T = (_Float16*)carve((size_t)112 * 224 * 2);
  _Float16* Wp3T = (_Float16*)carve((size_t)112 * 224 * 2);
  float*    F2   = (float*)carve((size_t)B * 30 * 200 * 4);
  (void)ws_size; (void)n_in; (void)out_size;

  // ---- dynamic LDS sizes ----
  const int SMEM1 = (2700 + 2700 + 32 + 32 + 30 * 260 + 30 * 258) * 4 + 32 * 1024 * 2;
  const int SMEM2 = POOLF * 4;
  hipFuncSetAttribute((const void*)conv_proj_kernel,
                      hipFuncAttributeMaxDynamicSharedMemorySize, SMEM1);
  hipFuncSetAttribute((const void*)graph_chain_kernel,
                      hipFuncAttributeMaxDynamicSharedMemorySize, SMEM2);

  // ---- weight prep (f16, zero-padded, transposed) ----
  auto pad = [&](const float* src, _Float16* dst, int K, int N, int Kp, int Np) {
    int total = Kp * Np;
    pad_cast_f16T<<<(total + 255) / 256, 256, 0, stream>>>(src, dst, K, N, Kp, Np);
  };
  pad(Wt,  WtT,  1024, 200, 1024, 208);
  pad(Wa,  WaT,  200, 200, 224, 208);
  pad(Wm1, Wm1T, 200, 200, 224, 208);
  pad(Wm2, Wm2T, 200, 200, 224, 208);
  pad(Wm3, Wm3T, 200, 200, 224, 208);
  pad(Wg1, Wg1T, 200, 200, 224, 208);
  pad(Wg2, Wg2T, 200, 200, 224, 208);
  pad(Wl,  WlT,  200, 150, 224, 160);
  pad(Wgl, WglT, 200, 150, 224, 160);
  pad(Wp1, Wp1T, 200, 100, 224, 112);
  pad(Wp2, Wp2T, 200, 100, 224, 112);
  pad(Wp3, Wp3T, 200, 100, 224, 112);

  // ---- stage 1: conv + temporal projection (WMMA) ----
  conv_proj_kernel<<<B, NTHREADS, SMEM1, stream>>>(x, Wc1, bc1, Wc2, bc2, WtT, bt, F2);

  // ---- stage 2: per-batch graph chain (WMMA) ----
  GraphArgs ga;
  ga.F2g = F2; ga.adj = adj;
  ga.Wa = WaT; ga.Wm1 = Wm1T; ga.Wm2 = Wm2T; ga.Wm3 = Wm3T;
  ga.Wg1 = Wg1T; ga.Wg2 = Wg2T; ga.Wl = WlT; ga.Wgl = WglT;
  ga.Wp0 = Wp1T; ga.Wp1 = Wp2T; ga.Wp2 = Wp3T;
  ga.wg = wg; ga.Ws1 = Ws1; ga.Ws2 = Ws2; ga.Wf1 = Wf1; ga.Wf2 = Wf2;
  ga.Wcls = Wcls; ga.bcls = bcls; ga.out = out;
  graph_chain_kernel<<<B, NTHREADS, SMEM2, stream>>>(ga);
}
